// KMeansClusteringLoss_1829656068283
// MI455X (gfx1250) — matrix operand order repaired and checked by reading, not compiled, not run on hardware
//
#include <hip/hip_runtime.h>
#include <hip/hip_bf16.h>

typedef __attribute__((ext_vector_type(16))) __bf16 bf16x16;
typedef __attribute__((ext_vector_type(8)))  __bf16 bf16x8;
typedef __attribute__((ext_vector_type(8)))  float  f32x8;
typedef int v4i __attribute__((vector_size(16)));

#define D_DIM         256
#define K_CENT        1024
#define ROWS_PER_BLK  128
#define WAVES_PER_BLK 8
#define CHUNK_COLS    64                      // centroids per streamed chunk
#define NCHUNK        (K_CENT / CHUNK_COLS)   // 16
#define LOADS_PER_CHUNK 8                     // async b128 per thread per chunk
#define XROW_STRIDE   260   // floats per x row in LDS (pad -> 4-bank rotation)
#define CROW_STRIDE   264   // bf16 per centroid row in LDS (pad)

static __device__ __forceinline__ unsigned short f2bf(float f) {
  unsigned u = __builtin_bit_cast(unsigned, f);
  u += 0x7FFFu + ((u >> 16) & 1u);            // round-to-nearest-even
  return (unsigned short)(u >> 16);
}

// ---- CDNA5 async global->LDS (ASYNCcnt path), 16B per lane ----------------
static __device__ __forceinline__ void async_b128(void* ldsPtr, const void* gPtr) {
#if __has_builtin(__builtin_amdgcn_global_load_async_to_lds_b128)
  __builtin_amdgcn_global_load_async_to_lds_b128(
      (__attribute__((address_space(1))) v4i*)gPtr,
      (__attribute__((address_space(3))) v4i*)ldsPtr, 0, 0);
#else
  unsigned l = (unsigned)(unsigned long)(__attribute__((address_space(3))) char*)ldsPtr;
  asm volatile("global_load_async_to_lds_b128 %0, %1, off"
               :: "v"(l), "v"(gPtr) : "memory");
#endif
}

static __device__ __forceinline__ void wait_async_le8() {
#if __has_builtin(__builtin_amdgcn_s_wait_asynccnt)
  __builtin_amdgcn_s_wait_asynccnt(8);
#else
  asm volatile("s_wait_asynccnt 0x8" ::: "memory");
#endif
}
static __device__ __forceinline__ void wait_async_0() {
#if __has_builtin(__builtin_amdgcn_s_wait_asynccnt)
  __builtin_amdgcn_s_wait_asynccnt(0);
#else
  asm volatile("s_wait_asynccnt 0x0" ::: "memory");
#endif
}

// B 32x16 bf16 fragment: lane = column; element i -> K = (i<8 ? i : i+8) + 8*lh (+32*kt)
static __device__ __forceinline__ bf16x16 loadB(const unsigned short* buf,
                                                int ct, int kt, int l16, int lh) {
  const __bf16* crow = (const __bf16*)(buf + (ct * 16 + l16) * CROW_STRIDE);
  const bf16x8* bp = (const bf16x8*)(crow + kt * 32 + lh * 8);
  return __builtin_shufflevector(bp[0], bp[2],
      0,1,2,3,4,5,6,7,8,9,10,11,12,13,14,15);
}

// ---------------------------------------------------------------------------
// Kernel A: centroids fp32 -> bf16 (workspace) + exact fp32 ||c||^2
// ---------------------------------------------------------------------------
__global__ void prep_centroids(const float* __restrict__ cent,
                               unsigned short* __restrict__ cbf,
                               float* __restrict__ csq) {
  const int lane = threadIdx.x & 31;
  const int wave = threadIdx.x >> 5;
  const int row  = blockIdx.x * WAVES_PER_BLK + wave;
  const float4* src = (const float4*)(cent + (size_t)row * D_DIM + lane * 8);
  float4 a = src[0], b = src[1];
  float ss = a.x*a.x + a.y*a.y + a.z*a.z + a.w*a.w
           + b.x*b.x + b.y*b.y + b.z*b.z + b.w*b.w;
  union { unsigned short h[8]; uint4 q; } pk;
  pk.h[0] = f2bf(a.x); pk.h[1] = f2bf(a.y); pk.h[2] = f2bf(a.z); pk.h[3] = f2bf(a.w);
  pk.h[4] = f2bf(b.x); pk.h[5] = f2bf(b.y); pk.h[6] = f2bf(b.z); pk.h[7] = f2bf(b.w);
  *(uint4*)(cbf + (size_t)row * D_DIM + lane * 8) = pk.q;
  for (int off = 16; off > 0; off >>= 1) ss += __shfl_xor(ss, off, 32);
  if (lane == 0) csq[row] = ss;
}

// ---------------------------------------------------------------------------
// Kernel B: 128 x-rows per block vs all 1024 centroids.
// ---------------------------------------------------------------------------
__global__ void __launch_bounds__(256, 2)
kmeans_min_kernel(const float* __restrict__ x,
                  const unsigned short* __restrict__ cbf,
                  const float* __restrict__ csq,
                  float* __restrict__ blockPartial) {
  extern __shared__ char smem[];
  float*  ldsX = (float*)smem;
  unsigned short* ldsC0 = (unsigned short*)smem;
  unsigned short* ldsC1 = (unsigned short*)(smem + CHUNK_COLS * CROW_STRIDE * 2);
  __shared__ float wavePartial[WAVES_PER_BLK];

  const int tid  = threadIdx.x;
  const int lane = tid & 31;
  const int wave = tid >> 5;
  const int lh   = lane >> 4;            // lane half (0: lanes 0-15, 1: 16-31)
  const int l16  = lane & 15;
  const size_t rowBase = (size_t)blockIdx.x * ROWS_PER_BLK;

  // ---- Phase 1a: async global -> LDS, fp32 x tile (32 x b128 per thread) ----
  {
    const char* gbase = (const char*)(x + rowBase * D_DIM);
    for (int it = 0; it < 32; ++it) {
      int idx = it * 256 + tid;          // float4 units, 64 per row
      int row = idx >> 6;
      int c4  = idx & 63;
      async_b128(ldsX + row * XROW_STRIDE + c4 * 4, gbase + row * 1024 + c4 * 16);
    }
    wait_async_0();
    __syncthreads();
  }

  // ---- Phase 1b: A fragments + x_sq ----
  // A 16x32 bf16 layout: lh=0 holds K {0..7,16..23}, lh=1 holds {8..15,24..31};
  // element i of v16bf -> K = (i<8 ? i : i+8) + 8*lh (+32*kt).
  bf16x16 aFrag[8];
  float xsq = 0.f;
  const float* xrow = ldsX + (wave * 16 + l16) * XROW_STRIDE;
#pragma unroll
  for (int kt = 0; kt < 8; ++kt) {
    const float4* p = (const float4*)(xrow + kt * 32 + lh * 8);
    float4 a0 = p[0], a1 = p[1];
    const float4* q = (const float4*)(xrow + kt * 32 + 16 + lh * 8);
    float4 b0 = q[0], b1 = q[1];
    xsq += a0.x*a0.x + a0.y*a0.y + a0.z*a0.z + a0.w*a0.w
         + a1.x*a1.x + a1.y*a1.y + a1.z*a1.z + a1.w*a1.w
         + b0.x*b0.x + b0.y*b0.y + b0.z*b0.z + b0.w*b0.w
         + b1.x*b1.x + b1.y*b1.y + b1.z*b1.z + b1.w*b1.w;
    union { unsigned short u[16]; bf16x16 v; } t;
    t.u[0] = f2bf(a0.x); t.u[1] = f2bf(a0.y); t.u[2]  = f2bf(a0.z); t.u[3]  = f2bf(a0.w);
    t.u[4] = f2bf(a1.x); t.u[5] = f2bf(a1.y); t.u[6]  = f2bf(a1.z); t.u[7]  = f2bf(a1.w);
    t.u[8] = f2bf(b0.x); t.u[9] = f2bf(b0.y); t.u[10] = f2bf(b0.z); t.u[11] = f2bf(b0.w);
    t.u[12]= f2bf(b1.x); t.u[13]= f2bf(b1.y); t.u[14] = f2bf(b1.z); t.u[15] = f2bf(b1.w);
    aFrag[kt] = t.v;
  }
  xsq += __shfl_xor(xsq, 16, 32);        // combine the two K-halves of the row
  float xsq_i[8];                        // x_sq for row M = i + 8*lh (C layout)
#pragma unroll
  for (int i = 0; i < 8; ++i) xsq_i[i] = __shfl(xsq, i + 8 * lh, 32);

  float minv[8];
#pragma unroll
  for (int i = 0; i < 8; ++i) minv[i] = 3.4e38f;

  __syncthreads();                       // all waves done reading x region

  // ---- Phase 2: double-buffered async centroid chunk stream ----
  const uint4* cbv = (const uint4*)cbf;
  for (int it = 0; it < LOADS_PER_CHUNK; ++it) {   // chunk 0 -> buffer 0
    int idx = it * 256 + tid;            // 16B units, 32 per centroid row
    int row = idx >> 5;
    int u   = idx & 31;
    async_b128(ldsC0 + row * CROW_STRIDE + u * 8,
               cbv + (size_t)row * (D_DIM / 8) + u);
  }

  for (int c = 0; c < NCHUNK; ++c) {
    unsigned short* bufCur = (c & 1) ? ldsC1 : ldsC0;
    unsigned short* bufNxt = (c & 1) ? ldsC0 : ldsC1;
    if (c + 1 < NCHUNK) {                // prefetch next chunk (async, in flight)
      for (int it = 0; it < LOADS_PER_CHUNK; ++it) {
        int idx = it * 256 + tid;
        int row = idx >> 5;
        int u   = idx & 31;
        async_b128(bufNxt + row * CROW_STRIDE + u * 8,
                   cbv + ((size_t)((c + 1) * CHUNK_COLS + row)) * (D_DIM / 8) + u);
      }
      wait_async_le8();                  // chunk c complete, chunk c+1 in flight
    } else {
      wait_async_0();
    }
    __syncthreads();                     // chunk c visible workgroup-wide

    float csv[CHUNK_COLS / 16];
#pragma unroll
    for (int ct = 0; ct < CHUNK_COLS / 16; ++ct)
      csv[ct] = csq[c * CHUNK_COLS + ct * 16 + l16];

    // 32-step flattened tile loop with 2-deep B-fragment prefetch pipeline
    bf16x16 bPipe0 = loadB(bufCur, 0, 0, l16, lh);
    bf16x16 bPipe1 = loadB(bufCur, 0, 1, l16, lh);
#pragma unroll
    for (int ct = 0; ct < CHUNK_COLS / 16; ++ct) {
      f32x8 acc = {0.f, 0.f, 0.f, 0.f, 0.f, 0.f, 0.f, 0.f};
#pragma unroll
      for (int kt = 0; kt < 8; ++kt) {
        const int n = ct * 8 + kt;
        bf16x16 bCur = (n & 1) ? bPipe1 : bPipe0;
        const int m = n + 2;             // issue load for step n+2 before WMMA n
        if (m < (CHUNK_COLS / 16) * 8) {
          if (m & 1) bPipe1 = loadB(bufCur, m >> 3, m & 7, l16, lh);
          else       bPipe0 = loadB(bufCur, m >> 3, m & 7, l16, lh);
        }
        acc = __builtin_amdgcn_wmma_f32_16x16x32_bf16(
            false, aFrag[kt], false, bCur, (short)0, acc, false, false);
      }
#pragma unroll
      for (int i = 0; i < 8; ++i) {
        float d2 = xsq_i[i] + csv[ct] - 2.f * acc[i];
        minv[i] = fminf(minv[i], d2);
      }
    }
    __syncthreads();                     // reads done before bufCur is rewritten
  }

  // ---- final min across the 16 lanes of each half (same row, diff cols) ----
  float s = 0.f;
#pragma unroll
  for (int i = 0; i < 8; ++i) {
    float m = minv[i];
    m = fminf(m, __shfl_xor(m, 1, 32));
    m = fminf(m, __shfl_xor(m, 2, 32));
    m = fminf(m, __shfl_xor(m, 4, 32));
    m = fminf(m, __shfl_xor(m, 8, 32));
    s += m;                              // identical across each 16-lane half
  }
  s += __shfl_xor(s, 16, 32);            // rows 0..7 half + rows 8..15 half
  if (lane == 0) wavePartial[wave] = s;
  __syncthreads();
  if (tid == 0) {
    float t = 0.f;
    for (int w = 0; w < WAVES_PER_BLK; ++w) t += wavePartial[w];
    blockPartial[blockIdx.x] = t;        // deterministic: no atomics
  }
}

// ---------------------------------------------------------------------------
// Kernel C: deterministic reduction of block partials -> mean
// ---------------------------------------------------------------------------
__global__ void finalize_mean(const float* __restrict__ partials, int nPart,
                              float invN, float* __restrict__ out) {
  __shared__ float red[256];
  float s = 0.f;
  for (int i = threadIdx.x; i < nPart; i += 256) s += partials[i];
  red[threadIdx.x] = s;
  __syncthreads();
  for (int st = 128; st > 0; st >>= 1) {
    if (threadIdx.x < st) red[threadIdx.x] += red[threadIdx.x + st];
    __syncthreads();
  }
  if (threadIdx.x == 0) out[0] = red[0] * invN;
}

extern "C" void kernel_launch(void* const* d_in, const int* in_sizes, int n_in,
                              void* d_out, int out_size, void* d_ws, size_t ws_size,
                              hipStream_t stream) {
  const float* x    = (const float*)d_in[0];
  const float* cent = (const float*)d_in[1];
  const int N = in_sizes[0] / D_DIM;               // 131072
  const int nBlocks = N / ROWS_PER_BLK;            // 1024

  // workspace layout: [bf16 centroids 512KB][c_sq 4KB][block partials 4KB]
  unsigned short* cbf = (unsigned short*)d_ws;
  float* csq      = (float*)((char*)d_ws + (size_t)K_CENT * D_DIM * 2);
  float* partials = csq + K_CENT;

  prep_centroids<<<K_CENT / WAVES_PER_BLK, 256, 0, stream>>>(cent, cbf, csq);

  const size_t smemBytes = (size_t)ROWS_PER_BLK * XROW_STRIDE * sizeof(float); // 133120
  kmeans_min_kernel<<<nBlocks, 256, smemBytes, stream>>>(x, cbf, csq, partials);

  finalize_mean<<<1, 256, 0, stream>>>(partials, nBlocks, 1.0f / (float)N,
                                       (float*)d_out);
}
